// Encoder_44298292691299
// MI455X (gfx1250) — compile-verified, matrix operand order checked by reading, not run on hardware
//
#include <hip/hip_runtime.h>
#include <hip/hip_bf16.h>
#include <math.h>

#define Bn 32
#define Tn 256
#define Hn 512
#define Gn 2048
#define Ln 2
#define RECUR_BLOCKS 8

typedef __attribute__((ext_vector_type(16))) __bf16 v16bf;
typedef __attribute__((ext_vector_type(16))) unsigned short v16u;
typedef __attribute__((ext_vector_type(8)))  float v8f;

__device__ __forceinline__ unsigned short f2bf(float f) {
  unsigned u = __builtin_bit_cast(unsigned, f);
  u += 0x7FFFu + ((u >> 16) & 1u);   // round-to-nearest-even
  return (unsigned short)(u >> 16);
}

// Load one lane's 16 bf16 fragment elements for a 16x32 WMMA tile.
// Lane half kb in {0,8}; halves 0..7 = K kb..kb+7, halves 8..15 = K kb+16..kb+23.
__device__ __forceinline__ v16bf load_tile16(const unsigned short* rowptr, int off) {
  union { uint4 q[2]; v16u v; } cv;
  const uint4* p = (const uint4*)(rowptr + off);
  cv.q[0] = p[0];          // K off .. off+7
  cv.q[1] = p[2];          // K off+16 .. off+23
  return __builtin_bit_cast(v16bf, cv.v);
}

__device__ __forceinline__ v8f wmma_bf16(v16bf a, v16bf b, v8f c) {
  return __builtin_amdgcn_wmma_f32_16x16x32_bf16(false, a, false, b, (short)0, c, false, false);
}

// fast sigmoid / tanh on v_exp_f32 + v_rcp_f32
__device__ __forceinline__ float sigm(float x) {
  return __builtin_amdgcn_rcpf(1.0f + __expf(-x));
}
__device__ __forceinline__ float tanh_fast(float x) {
  return 1.0f - 2.0f * __builtin_amdgcn_rcpf(1.0f + __expf(2.0f * x));
}

__global__ void __launch_bounds__(256) embed_cast_kernel(
    const int* __restrict__ tokens, const float* __restrict__ emb,
    unsigned short* __restrict__ xbf) {
  int idx = blockIdx.x * 256 + threadIdx.x;     // [0, Bn*Tn*Hn)
  int row = idx >> 9;                           // token position (b*T + t)
  int col = idx & (Hn - 1);
  int tok = tokens[row];
  xbf[idx] = f2bf(emb[tok * Hn + col]);
}

// w: [Hn, Gn] row-major (K x N)  ->  wt: [Gn, Hn] (N x K), bf16
__global__ void __launch_bounds__(256) cast_transpose_kernel(
    const float* __restrict__ w, unsigned short* __restrict__ wt) {
  int idx = blockIdx.x * 256 + threadIdx.x;     // [0, Hn*Gn)
  int k = idx >> 11;                            // / Gn
  int n = idx & (Gn - 1);
  wt[n * Hn + k] = f2bf(w[idx]);
}

__global__ void __launch_bounds__(64) zero_bar_kernel(unsigned int* bar) {
  bar[threadIdx.x] = 0u;
}

// Zx = Xbf @ WxT + bias   (M = 8192, N = 2048, K = 512), bf16 WMMA, f32 accum
__global__ void __launch_bounds__(256) gemm_zx_kernel(
    const unsigned short* __restrict__ xbf,   // [8192, 512] bf16
    const unsigned short* __restrict__ wxt,   // [2048, 512] bf16 (transposed W)
    const float* __restrict__ bias,           // [2048]
    float* __restrict__ zx) {                 // [8192, 2048] f32
  const int lane  = threadIdx.x & 31;
  const int w     = blockIdx.x * 8 + (threadIdx.x >> 5); // 0..16383 wave tasks
  const int mtile = w & 511;
  const int ngrp  = w >> 9;                   // 0..31  (64 columns per wave)
  const int kb    = (lane >> 4) << 3;
  const int mrow  = mtile * 16 + (lane & 15);
  const unsigned short* arow = xbf + mrow * Hn;
  const unsigned short* brow[4];
  int col[4];
  #pragma unroll
  for (int g = 0; g < 4; ++g) {
    col[g]  = ngrp * 64 + g * 16 + (lane & 15);
    brow[g] = wxt + col[g] * Hn;
  }

  v8f acc[4] = {};
  // software pipeline: prefetch k+1 fragments while issuing k's WMMAs
  v16bf a_c = load_tile16(arow, kb);
  v16bf b_c[4];
  #pragma unroll
  for (int g = 0; g < 4; ++g) b_c[g] = load_tile16(brow[g], kb);

  #pragma unroll 4
  for (int kk = 0; kk < 16; ++kk) {
    const int offn = ((kk + 1) & 15) * 32 + kb;     // wraps (harmless reload) on last iter
    v16bf a_n  = load_tile16(arow, offn);
    v16bf b_n0 = load_tile16(brow[0], offn);
    v16bf b_n1 = load_tile16(brow[1], offn);
    v16bf b_n2 = load_tile16(brow[2], offn);
    v16bf b_n3 = load_tile16(brow[3], offn);
    acc[0] = wmma_bf16(a_c, b_c[0], acc[0]);
    acc[1] = wmma_bf16(a_c, b_c[1], acc[1]);
    acc[2] = wmma_bf16(a_c, b_c[2], acc[2]);
    acc[3] = wmma_bf16(a_c, b_c[3], acc[3]);
    a_c = a_n; b_c[0] = b_n0; b_c[1] = b_n1; b_c[2] = b_n2; b_c[3] = b_n3;
  }

  #pragma unroll
  for (int g = 0; g < 4; ++g) {
    float bv = bias[col[g]];
    #pragma unroll
    for (int r = 0; r < 8; ++r) {
      int m = mtile * 16 + r + ((lane >> 4) << 3);   // C layout: M = r + 8*(lane>=16)
      zx[m * Gn + col[g]] = acc[g][r] + bv;
    }
  }
}

__device__ __forceinline__ void grid_sync(unsigned int* bar) {
  __syncthreads();
  if (threadIdx.x == 0) {
    unsigned int gen = __hip_atomic_load(&bar[1], __ATOMIC_ACQUIRE, __HIP_MEMORY_SCOPE_AGENT);
    unsigned int arrived =
        __hip_atomic_fetch_add(&bar[0], 1u, __ATOMIC_ACQ_REL, __HIP_MEMORY_SCOPE_AGENT) + 1u;
    if (arrived == (unsigned int)gridDim.x) {
      __hip_atomic_store(&bar[0], 0u, __ATOMIC_RELAXED, __HIP_MEMORY_SCOPE_AGENT);
      __hip_atomic_fetch_add(&bar[1], 1u, __ATOMIC_ACQ_REL, __HIP_MEMORY_SCOPE_AGENT);
    } else {
      while (__hip_atomic_load(&bar[1], __ATOMIC_ACQUIRE, __HIP_MEMORY_SCOPE_AGENT) == gen) {
        __builtin_amdgcn_s_sleep(1);
      }
    }
  }
  __syncthreads();
}

// Persistent recurrence: 8 blocks x 256 threads = 64 waves.
// wave -> (mtile in 0..1, jt in 0..31): owns batches mtile*16..+15, columns jt*16..+15, all 4 gates.
__global__ void __launch_bounds__(256) lstm_recur_kernel(
    const float* __restrict__ zx,            // [8192, 2048] precomputed x@Wx + b
    const unsigned short* __restrict__ wht,  // [2048, 512] bf16 transposed Wh
    const int* __restrict__ lengths,         // [32]
    unsigned short* __restrict__ hbf,        // [2][32,512] bf16 double-buffered hidden
    unsigned short* __restrict__ xbf_next,   // [8192,512] next layer input (bf16)
    int write_next,
    float* __restrict__ out_x,               // [32,256,512]
    float* __restrict__ out_c,               // [32,512]
    float* __restrict__ out_h,               // [32,512]
    unsigned int* bar) {
  const int tid   = threadIdx.x;
  const int lane  = tid & 31;
  const int w     = blockIdx.x * 8 + (tid >> 5);  // 0..63
  const int mtile = w & 1;
  const int jt    = w >> 1;                       // 0..31
  const int kb    = (lane >> 4) << 3;
  const int nj    = jt * 16 + (lane & 15);        // hidden column 0..511

  unsigned short* hbuf0 = hbf;
  unsigned short* hbuf1 = hbf + Bn * Hn;

  // zero read-buffer for t=0
  for (int i = blockIdx.x * 256 + tid; i < Bn * Hn; i += RECUR_BLOCKS * 256) hbuf0[i] = 0;

  const unsigned short* brow[4];
  #pragma unroll
  for (int g = 0; g < 4; ++g) brow[g] = wht + (g * Hn + nj) * Hn;

  int brows[8]; int lenb[8]; int zoff[8]; int ooff[8]; int hoff[8];
  #pragma unroll
  for (int r = 0; r < 8; ++r) {
    brows[r] = mtile * 16 + r + ((lane >> 4) << 3);
    lenb[r]  = lengths[brows[r]];
    zoff[r]  = brows[r] * Tn * Gn + nj;     // advances by Gn per step
    ooff[r]  = brows[r] * Tn * Hn + nj;     // advances by Hn per step
    hoff[r]  = brows[r] * Hn + nj;          // constant
  }
  float cst[8] = {0.f,0.f,0.f,0.f,0.f,0.f,0.f,0.f};
  float hst[8] = {0.f,0.f,0.f,0.f,0.f,0.f,0.f,0.f};

  grid_sync(bar);

  for (int t = 0; t < Tn; ++t) {
    const unsigned short* hr = (t & 1) ? hbuf1 : hbuf0;
    unsigned short*       hw = (t & 1) ? hbuf0 : hbuf1;
    const unsigned short* arow = hr + (mtile * 16 + (lane & 15)) * Hn;

    // early-issue the gate pre-activation loads (independent of h)
    float z0[8], z1[8], z2[8], z3[8];
    #pragma unroll
    for (int r = 0; r < 8; ++r) {
      z0[r] = zx[zoff[r]];
      z1[r] = zx[zoff[r] + Hn];
      z2[r] = zx[zoff[r] + 2 * Hn];
      z3[r] = zx[zoff[r] + 3 * Hn];
    }

    v8f acc[4] = {};
    v16bf a_c = load_tile16(arow, kb);
    v16bf b_c[4];
    #pragma unroll
    for (int g = 0; g < 4; ++g) b_c[g] = load_tile16(brow[g], kb);

    #pragma unroll 4
    for (int kk = 0; kk < 16; ++kk) {
      const int offn = ((kk + 1) & 15) * 32 + kb;
      v16bf a_n  = load_tile16(arow, offn);
      v16bf b_n0 = load_tile16(brow[0], offn);
      v16bf b_n1 = load_tile16(brow[1], offn);
      v16bf b_n2 = load_tile16(brow[2], offn);
      v16bf b_n3 = load_tile16(brow[3], offn);
      acc[0] = wmma_bf16(a_c, b_c[0], acc[0]);
      acc[1] = wmma_bf16(a_c, b_c[1], acc[1]);
      acc[2] = wmma_bf16(a_c, b_c[2], acc[2]);
      acc[3] = wmma_bf16(a_c, b_c[3], acc[3]);
      a_c = a_n; b_c[0] = b_n0; b_c[1] = b_n1; b_c[2] = b_n2; b_c[3] = b_n3;
    }

    #pragma unroll
    for (int r = 0; r < 8; ++r) {
      float zi = acc[0][r] + z0[r];
      float zf = acc[1][r] + z1[r];
      float zg = acc[2][r] + z2[r];
      float zo = acc[3][r] + z3[r];
      float nc = sigm(zf) * cst[r] + sigm(zi) * tanh_fast(zg);
      float nh = sigm(zo) * tanh_fast(nc);
      bool live = (t < lenb[r]);
      cst[r] = live ? nc : cst[r];
      hst[r] = live ? nh : hst[r];
      out_x[ooff[r]] = hst[r];
      unsigned short hb = f2bf(hst[r]);
      hw[hoff[r]] = hb;
      if (write_next) xbf_next[ooff[r]] = hb;
      zoff[r] += Gn;
      ooff[r] += Hn;
    }
    __threadfence();
    grid_sync(bar);
  }
  #pragma unroll
  for (int r = 0; r < 8; ++r) {
    out_c[hoff[r]] = cst[r];
    out_h[hoff[r]] = hst[r];
  }
}

extern "C" void kernel_launch(void* const* d_in, const int* in_sizes, int n_in,
                              void* d_out, int out_size, void* d_ws, size_t ws_size,
                              hipStream_t stream) {
  const int*   tokens  = (const int*)d_in[0];
  const int*   lengths = (const int*)d_in[1];
  const float* emb     = (const float*)d_in[2];
  const float* Wx      = (const float*)d_in[3];
  const float* Wh      = (const float*)d_in[4];
  const float* bias    = (const float*)d_in[5];

  char* p = (char*)d_ws;
  unsigned short* xbf = (unsigned short*)p; p += (size_t)Bn * Tn * Hn * 2;  //  8 MB
  unsigned short* wxt = (unsigned short*)p; p += (size_t)Ln * Hn * Gn * 2;  //  4 MB
  unsigned short* wht = (unsigned short*)p; p += (size_t)Ln * Hn * Gn * 2;  //  4 MB
  unsigned short* hbf = (unsigned short*)p; p += (size_t)2 * Bn * Hn * 2;   // 64 KB
  unsigned int*   bar = (unsigned int*)p;   p += 256;
  float*          zx  = (float*)p;          p += (size_t)Bn * Tn * Gn * 4;  // 64 MB

  float* out_x = (float*)d_out;
  float* out_c = out_x + (size_t)Bn * Tn * Hn;
  float* out_h = out_c + (size_t)Bn * Hn;

  zero_bar_kernel<<<1, 64, 0, stream>>>(bar);
  embed_cast_kernel<<<(Bn * Tn * Hn) / 256, 256, 0, stream>>>(tokens, emb, xbf);
  for (int l = 0; l < Ln; ++l) {
    cast_transpose_kernel<<<(Hn * Gn) / 256, 256, 0, stream>>>(
        Wx + (size_t)l * Hn * Gn, wxt + (size_t)l * Gn * Hn);
    cast_transpose_kernel<<<(Hn * Gn) / 256, 256, 0, stream>>>(
        Wh + (size_t)l * Hn * Gn, wht + (size_t)l * Gn * Hn);
  }
  for (int l = 0; l < Ln; ++l) {
    gemm_zx_kernel<<<2048, 256, 0, stream>>>(
        xbf, wxt + (size_t)l * Gn * Hn, bias + l * Gn, zx);
    lstm_recur_kernel<<<RECUR_BLOCKS, 256, 0, stream>>>(
        zx, wht + (size_t)l * Gn * Hn, lengths, hbf, xbf,
        (l + 1 < Ln) ? 1 : 0, out_x, out_c, out_h, bar);
  }
}
